// MyModel_59365037965464
// MI455X (gfx1250) — matrix operand order repaired and checked by reading, not compiled, not run on hardware
//
#include <hip/hip_runtime.h>
#include <math.h>

typedef _Float16 half16 __attribute__((ext_vector_type(16)));
typedef _Float16 half8  __attribute__((ext_vector_type(8)));
typedef float    f32x8  __attribute__((ext_vector_type(8)));

#define N_ROWS   12000
#define D_INX    129
#define DF       128
#define HS       256
#define KNBR     5
#define THRESH   0.98f
#define HIST_MIN 1000
#define XI_LD    288   // 257 padded to multiple of 32
#define XC_LD    384   // 256 + 128
#define CAP      24    // max stored candidates (> 0.98) per row
#define JC       1024  // j-chunk width per wave in sim scan

// A-operand chunk (f16 16x16x32 WMMA): two aligned 16B segments per lane.
static __device__ __forceinline__ half16 load_a_chunk(const _Float16* p) {
    half8 lo = *(const half8*)p;          // K 0..7   (or 16..23 for upper half-wave)
    half8 hi = *(const half8*)(p + 8);    // K 8..15  (or 24..31)
    return __builtin_shufflevector(lo, hi, 0, 1, 2, 3, 4, 5, 6, 7,
                                           8, 9, 10, 11, 12, 13, 14, 15);
}
// B-operand chunk: one aligned 32B segment per lane.
static __device__ __forceinline__ half16 load_b_chunk(const _Float16* p) {
    return *(const half16*)p;
}

// ---------------------------------------------------------------------------
// Kernel 1: price + xf(f16) + row-normalized xn(f16).  One wave per row.
// ---------------------------------------------------------------------------
__global__ void prep_x(const float* __restrict__ x, _Float16* __restrict__ xf16,
                       _Float16* __restrict__ xn16, float* __restrict__ price) {
    int wave = (blockIdx.x * blockDim.x + threadIdx.x) >> 5;
    int lane = threadIdx.x & 31;
    if (wave >= N_ROWS) return;
    const float* row = x + (long)wave * D_INX;
    float v[4];
    float ss = 0.f;
#pragma unroll
    for (int t = 0; t < 4; ++t) { v[t] = row[1 + lane * 4 + t]; ss += v[t] * v[t]; }
#pragma unroll
    for (int off = 16; off > 0; off >>= 1) ss += __shfl_xor(ss, off, 32);
    float inv = rsqrtf(ss);
    _Float16* pf = xf16 + (long)wave * DF + lane * 4;
    _Float16* pn = xn16 + (long)wave * DF + lane * 4;
#pragma unroll
    for (int t = 0; t < 4; ++t) { pf[t] = (_Float16)v[t]; pn[t] = (_Float16)(v[t] * inv); }
    if (lane == 0) price[wave] = row[0];
}

// ---------------------------------------------------------------------------
// Kernel 2: f32 -> f16 weight conversion with zero-padding of K dimension.
// ---------------------------------------------------------------------------
__global__ void convert_pad(const float* __restrict__ src, _Float16* __restrict__ dst,
                            int rows, int sc, int dc) {
    int idx = blockIdx.x * blockDim.x + threadIdx.x;
    if (idx >= rows * dc) return;
    int r = idx / dc, c = idx % dc;
    dst[idx] = (c < sc) ? (_Float16)src[r * sc + c] : (_Float16)0.f;
}

__global__ void zero_i32(int* __restrict__ p, int n) {
    int i = blockIdx.x * blockDim.x + threadIdx.x;
    if (i < n) p[i] = 0;
}

// ---------------------------------------------------------------------------
// Kernel 3: generic WMMA GEMM  C = act(A @ W^T + bias).
// A: MxK f16 (lda), W: NoutxK f16 (ldw), K % 32 == 0, Nout % 64 == 0.
// One wave computes a 16x64 tile: A chunk loaded once per K-step, reused
// across 4 v_wmma_f32_16x16x32_f16. ACT: 0 none, 1 tanh, 2 relu.
// ---------------------------------------------------------------------------
template <int ACT>
__global__ __launch_bounds__(128, 1) void
wmma_gemm(const _Float16* __restrict__ A, int lda,
          const _Float16* __restrict__ W, int ldw,
          const float* __restrict__ bias,
          int M, int Nout, int Kdim,
          float* __restrict__ outF32, int ldo32,
          _Float16* __restrict__ outF16, int ldo16) {
    int wid  = blockIdx.x * (blockDim.x >> 5) + (threadIdx.x >> 5);
    int lane = threadIdx.x & 31;
    int tilesN = Nout >> 6;                  // 64 output cols per wave
    int tileM = wid / tilesN, tN = wid % tilesN;
    if (tileM * 16 >= M) return;
    int half_ = lane >> 4;
    int l16   = lane & 15;
    const _Float16* arow = A + (long)(tileM * 16 + l16) * lda + (half_ ? 16 : 0);
    const _Float16* wrow = W + (long)(tN * 64 + l16) * ldw + (half_ ? 16 : 0);

    f32x8 acc[4] = {};
    for (int kc = 0; kc < Kdim; kc += 32) {
        half16 a = load_a_chunk(arow + kc);
        half16 b[4];
#pragma unroll
        for (int nt = 0; nt < 4; ++nt)
            b[nt] = load_b_chunk(wrow + (long)nt * 16 * ldw + kc);
#pragma unroll
        for (int nt = 0; nt < 4; ++nt)
            acc[nt] = __builtin_amdgcn_wmma_f32_16x16x32_f16(false, a, false, b[nt],
                                                             (short)0, acc[nt], false, false);
    }
#pragma unroll
    for (int nt = 0; nt < 4; ++nt) {
        int col = tN * 64 + nt * 16 + l16;
        float bv = bias ? bias[col] : 0.f;
#pragma unroll
        for (int v = 0; v < 8; ++v) {
            int row = tileM * 16 + v + 8 * half_;
            float val = acc[nt][v] + bv;
            if (ACT == 1) val = tanhf(val);
            if (ACT == 2) val = fmaxf(val, 0.f);
            if (outF32) outF32[(long)row * ldo32 + col] = val;
            if (outF16) outF16[(long)row * ldo16 + col] = (_Float16)val;
        }
    }
}

// ---------------------------------------------------------------------------
// Kernel 4: similarity scan.  Wave owns 32 rows x one 1024-col j-chunk.
// Per 16-col tile: 8 B loads (all in flight) -> 8 WMMA -> v_max tree + __any.
// Rare hits (sim > 0.98) append (j, sim) to per-row candidate lists with
// exact per-row column-validity: j < max(1000,i)  <=>  (j-i < 0) || (j < 1000).
// ---------------------------------------------------------------------------
__global__ __launch_bounds__(128, 1) void
sim_scan(const _Float16* __restrict__ xn16,
         int* __restrict__ candCnt, int* __restrict__ candJ,
         float* __restrict__ candV) {
    int wv   = threadIdx.x >> 5;
    int lane = threadIdx.x & 31;
    int r0   = blockIdx.x * 32;                       // 32-row block
    int half_ = lane >> 4, l16 = lane & 15;

    int jlimit = (r0 + 31 > HIST_MIN) ? (r0 + 31) : HIST_MIN;
    int jstart = (blockIdx.y * 4 + wv) * JC;
    if (jstart >= jlimit) return;
    int jend = (jstart + JC < jlimit) ? (jstart + JC) : jlimit;

    // A operand: rows r0..r0+31 (two 16-row tiles), K=128 -> 4 chunks each.
    half16 a0[4], a1[4];
    const _Float16* ab0 = xn16 + (long)(r0 + l16) * DF + (half_ ? 16 : 0);
    const _Float16* ab1 = ab0 + 16 * DF;
#pragma unroll
    for (int kc = 0; kc < 4; ++kc) {
        a0[kc] = load_a_chunk(ab0 + kc * 32);
        a1[kc] = load_a_chunk(ab1 + kc * 32);
    }

    for (int j0 = jstart; j0 < jend; j0 += 16) {
        const _Float16* bbase = xn16 + (long)(j0 + l16) * DF + (half_ ? 16 : 0);
        half16 bc[4];
#pragma unroll
        for (int kc = 0; kc < 4; ++kc) bc[kc] = load_b_chunk(bbase + kc * 32);
        if (j0 + 16 < jend) {   // prefetch next B tile (global_prefetch_b8)
            const char* np = (const char*)(xn16 + (long)(j0 + 16 + l16) * DF);
            __builtin_prefetch(np, 0, 1);
            __builtin_prefetch(np + 128, 0, 1);
        }
        f32x8 c0 = {}, c1 = {};
#pragma unroll
        for (int kc = 0; kc < 4; ++kc) {
            c0 = __builtin_amdgcn_wmma_f32_16x16x32_f16(false, a0[kc], false, bc[kc],
                                                        (short)0, c0, false, false);
            c1 = __builtin_amdgcn_wmma_f32_16x16x32_f16(false, a1[kc], false, bc[kc],
                                                        (short)0, c1, false, false);
        }
        // fast check: any value anywhere above threshold?
        float mx = c0[0];
#pragma unroll
        for (int v = 1; v < 8; ++v) mx = fmaxf(mx, c0[v]);
#pragma unroll
        for (int v = 0; v < 8; ++v) mx = fmaxf(mx, c1[v]);
        if (__any(mx > THRESH)) {             // rare: duplicates / diagonal tile
            int j    = j0 + l16;
            bool jlt = j < HIST_MIN;
            int ib0  = r0 + 8 * half_;        // i = ib0 + v (tile0), ib0 + 16 + v (tile1)
            int db0  = j - ib0;               // j - i = db0 - v
            int db1  = db0 - 16;
#pragma unroll
            for (int v = 0; v < 8; ++v) {
                if (c0[v] > THRESH && (db0 < v || jlt)) {
                    int i = ib0 + v;
                    int slot = atomicAdd(&candCnt[i], 1);
                    if (slot < CAP) { candJ[i * CAP + slot] = j; candV[i * CAP + slot] = c0[v]; }
                }
                if (c1[v] > THRESH && (db1 < v || jlt)) {
                    int i = ib0 + 16 + v;
                    int slot = atomicAdd(&candCnt[i], 1);
                    if (slot < CAP) { candJ[i * CAP + slot] = j; candV[i * CAP + slot] = c1[v]; }
                }
            }
        }
    }
}

// ---------------------------------------------------------------------------
// Kernel 5: exact top-5 over each row's candidate list (value desc, index
// tie-break => deterministic), then mask / count / mean neighbor price.
// ---------------------------------------------------------------------------
__global__ void topk_finalize(const int* __restrict__ candCnt, const int* __restrict__ candJ,
                              const float* __restrict__ candV, const float* __restrict__ price,
                              int* __restrict__ nbr, float* __restrict__ maskf,
                              float* __restrict__ cntf, float* __restrict__ xprice1) {
    int i = blockIdx.x * blockDim.x + threadIdx.x;
    if (i >= N_ROWS) return;
    int cc = candCnt[i];
    if (cc > CAP) cc = CAP;
    float tv[KNBR]; int ti[KNBR];
#pragma unroll
    for (int s = 0; s < KNBR; ++s) { tv[s] = -3.0e38f; ti[s] = 0x7fffffff; }
    for (int k = 0; k < cc; ++k) {
        float cv = candV[i * CAP + k];
        int   cj = candJ[i * CAP + k];
#pragma unroll
        for (int s = 0; s < KNBR; ++s) {     // descending insertion, low-index wins ties
            bool gt = (cv > tv[s]) || (cv == tv[s] && cj < ti[s]);
            float ov = tv[s]; int oi = ti[s];
            tv[s] = gt ? cv : ov;  ti[s] = gt ? cj : oi;
            cv    = gt ? ov : cv;  cj    = gt ? oi : cj;
        }
    }
    float cnt = 0.f, psum = 0.f;
#pragma unroll
    for (int s = 0; s < KNBR; ++s) {
        bool m = (tv[s] > THRESH) && (ti[s] != i);
        float mf = m ? 1.f : 0.f;
        cnt += mf;
        if (m) psum += price[ti[s]];
        nbr[i * KNBR + s]   = m ? ti[s] : 0;
        maskf[i * KNBR + s] = mf;
    }
    cntf[i]    = cnt;
    xprice1[i] = psum / fmaxf(cnt, 1.f);
}

// ---------------------------------------------------------------------------
// Kernel 6: place x_price1 into x_i (col 256) and zero the K padding.
// ---------------------------------------------------------------------------
__global__ void scatter_xi(const float* __restrict__ xprice1, _Float16* __restrict__ xi16) {
    int i = blockIdx.x * blockDim.x + threadIdx.x;
    if (i >= N_ROWS) return;
    _Float16* p = xi16 + (long)i * XI_LD;
    p[256] = (_Float16)xprice1[i];
#pragma unroll
    for (int c = 257; c < XI_LD; ++c) p[c] = (_Float16)0.f;
}

// ---------------------------------------------------------------------------
// Kernel 7: x_conv = (Wl @ sum_masked(x_i[nbr]-x_j) + cnt*bl) / max(cnt,1).
// One wave per node; summed diff staged in LDS; writes xcat cols 256..383.
// ---------------------------------------------------------------------------
__global__ void msg_conv(const float* __restrict__ h, const float* __restrict__ price,
                         const float* __restrict__ xprice1,
                         const int* __restrict__ nbr, const float* __restrict__ maskf,
                         const float* __restrict__ cntf,
                         const float* __restrict__ Wl, const float* __restrict__ bl,
                         _Float16* __restrict__ xcat) {
    __shared__ float sdiff[4][257];
    int wv = threadIdx.x >> 5, lane = threadIdx.x & 31;
    int i = blockIdx.x * 4 + wv;
    if (i >= N_ROWS) return;
    float cnt = cntf[i];

    for (int d = lane; d < 257; d += 32) {
        float s = 0.f;
#pragma unroll
        for (int t = 0; t < KNBR; ++t) {
            float mf = maskf[i * KNBR + t];
            if (mf > 0.f) {
                int nb = nbr[i * KNBR + t];
                float xi_d = (d < 256) ? h[(long)nb * HS + d] : xprice1[nb];
                float xj_d = (d < 256) ? h[(long)i * HS + d] : price[i];
                s += xi_d - xj_d;
            }
        }
        sdiff[wv][d] = s;
    }
    asm volatile("s_wait_dscnt 0x0" ::: "memory");
    __builtin_amdgcn_wave_barrier();

    float safe = fmaxf(cnt, 1.f);
    float acc[4] = {0.f, 0.f, 0.f, 0.f};
    for (int d = 0; d < 257; ++d) {
        float v = sdiff[wv][d];
#pragma unroll
        for (int t = 0; t < 4; ++t)
            acc[t] = fmaf(Wl[(long)(lane + 32 * t) * 257 + d], v, acc[t]);
    }
    _Float16* outp = xcat + (long)i * XC_LD + 256;
#pragma unroll
    for (int t = 0; t < 4; ++t) {
        int c = lane + 32 * t;
        outp[c] = (_Float16)((acc[t] + cnt * bl[c]) / safe);
    }
}

// ---------------------------------------------------------------------------
// Kernel 8: out = x3 @ W3^T + b3.  One wave per node, shuffle reduction.
// ---------------------------------------------------------------------------
__global__ void final_out(const float* __restrict__ x3, const float* __restrict__ W3,
                          const float* __restrict__ b3, float* __restrict__ out) {
    int wv = threadIdx.x >> 5, lane = threadIdx.x & 31;
    int i = blockIdx.x * 4 + wv;
    if (i >= N_ROWS) return;
    float s = 0.f;
#pragma unroll
    for (int d = lane; d < HS; d += 32) s = fmaf(x3[(long)i * HS + d], W3[d], s);
#pragma unroll
    for (int off = 16; off > 0; off >>= 1) s += __shfl_xor(s, off, 32);
    if (lane == 0) out[i] = s + b3[0];
}

// ---------------------------------------------------------------------------
extern "C" void kernel_launch(void* const* d_in, const int* in_sizes, int n_in,
                              void* d_out, int out_size, void* d_ws, size_t ws_size,
                              hipStream_t stream) {
    (void)in_sizes; (void)n_in; (void)out_size; (void)ws_size;
    const float* x   = (const float*)d_in[0];
    const float* W1  = (const float*)d_in[1];
    const float* b1  = (const float*)d_in[2];
    const float* Wl  = (const float*)d_in[3];
    const float* bl  = (const float*)d_in[4];
    const float* W2  = (const float*)d_in[5];
    const float* b2  = (const float*)d_in[6];
    const float* W21 = (const float*)d_in[7];
    const float* b21 = (const float*)d_in[8];
    const float* W3  = (const float*)d_in[9];
    const float* b3  = (const float*)d_in[10];
    float* out = (float*)d_out;

    char* ws = (char*)d_ws;
    size_t off = 0;
    auto alloc = [&](size_t bytes) -> void* {
        void* p = ws + off;
        off = (off + bytes + 255) & ~(size_t)255;
        return p;
    };
    _Float16* xf16   = (_Float16*)alloc((size_t)N_ROWS * DF * 2);
    _Float16* xn16   = (_Float16*)alloc((size_t)N_ROWS * DF * 2);
    float*    price  = (float*)   alloc((size_t)N_ROWS * 4);
    float*    h32    = (float*)   alloc((size_t)N_ROWS * HS * 4);
    _Float16* xi16   = (_Float16*)alloc((size_t)N_ROWS * XI_LD * 2);
    _Float16* xcat16 = (_Float16*)alloc((size_t)N_ROWS * XC_LD * 2);
    float*    x3f    = (float*)   alloc((size_t)N_ROWS * HS * 4);
    float*    xp1    = (float*)   alloc((size_t)N_ROWS * 4);
    float*    cnt    = (float*)   alloc((size_t)N_ROWS * 4);
    float*    mskf   = (float*)   alloc((size_t)N_ROWS * KNBR * 4);
    int*      nbri   = (int*)     alloc((size_t)N_ROWS * KNBR * 4);
    int*      candC  = (int*)     alloc((size_t)N_ROWS * 4);
    int*      candJ  = (int*)     alloc((size_t)N_ROWS * CAP * 4);
    float*    candV  = (float*)   alloc((size_t)N_ROWS * CAP * 4);
    _Float16* W1h    = (_Float16*)alloc((size_t)HS * DF * 2);
    _Float16* W2h    = (_Float16*)alloc((size_t)HS * XI_LD * 2);
    _Float16* W21h   = (_Float16*)alloc((size_t)HS * XC_LD * 2);

    // weights -> f16 (with K zero-padding); reset candidate counters
    convert_pad<<<(HS * DF + 255) / 256, 256, 0, stream>>>(W1, W1h, HS, DF, DF);
    convert_pad<<<(HS * XI_LD + 255) / 256, 256, 0, stream>>>(W2, W2h, HS, 257, XI_LD);
    convert_pad<<<(HS * XC_LD + 255) / 256, 256, 0, stream>>>(W21, W21h, HS, XC_LD, XC_LD);
    zero_i32<<<(N_ROWS + 255) / 256, 256, 0, stream>>>(candC, N_ROWS);

    // price / xf / xn
    prep_x<<<N_ROWS / 4, 128, 0, stream>>>(x, xf16, xn16, price);

    // h = tanh(xf @ W1^T + b1): f32 copy + f16 copy into x_i cols 0..255
    int gemm_blocks = (N_ROWS / 16) * (HS / 64) / 4;   // 4 waves/block, 16x64 per wave
    wmma_gemm<1><<<gemm_blocks, 128, 0, stream>>>(xf16, DF, W1h, DF, b1,
                                                  N_ROWS, HS, DF, h32, HS, xi16, XI_LD);

    // similarity scan: 375 row-blocks x 12 j-chunks (4 chunks per block)
    dim3 simGrid(N_ROWS / 32, 3);
    sim_scan<<<simGrid, 128, 0, stream>>>(xn16, candC, candJ, candV);
    topk_finalize<<<(N_ROWS + 255) / 256, 256, 0, stream>>>(candC, candJ, candV, price,
                                                            nbri, mskf, cnt, xp1);

    // x_i[:,256] = x_price1, zero padding
    scatter_xi<<<(N_ROWS + 255) / 256, 256, 0, stream>>>(xp1, xi16);

    // x2 = relu(x_i @ W2^T + b2) -> xcat cols 0..255 (f16)
    wmma_gemm<2><<<gemm_blocks, 128, 0, stream>>>(xi16, XI_LD, W2h, XI_LD, b2,
                                                  N_ROWS, HS, XI_LD, nullptr, 0, xcat16, XC_LD);

    // x_conv -> xcat cols 256..383 (f16)
    msg_conv<<<N_ROWS / 4, 128, 0, stream>>>(h32, price, xp1, nbri, mskf, cnt, Wl, bl, xcat16);

    // x3 = relu(xcat @ W21^T + b21) (f32)
    wmma_gemm<2><<<gemm_blocks, 128, 0, stream>>>(xcat16, XC_LD, W21h, XC_LD, b21,
                                                  N_ROWS, HS, XC_LD, x3f, HS, nullptr, 0);

    // out = x3 @ W3^T + b3
    final_out<<<N_ROWS / 4, 128, 0, stream>>>(x3f, W3, b3, out);
}
